// MolecularGraphNet_44839458570700
// MI455X (gfx1250) — matrix operand order repaired and checked by reading, not compiled, not run on hardware
//
#include <hip/hip_runtime.h>
#include <cstdint>

// ---------------------------------------------------------------------------
// GraphSAGE (3 layers, mean aggregation) + FC for MI455X / gfx1250.
// GEMMs use V_WMMA_F32_16X16X32_BF16 (bf16 A/B, f32 accumulate).
// Scatter/aggregate uses global_atomic_add_f32 streaming (bandwidth bound).
// ---------------------------------------------------------------------------

typedef __bf16 bf16_t;
typedef bf16_t v16bf __attribute__((ext_vector_type(16)));
typedef float  v8f   __attribute__((ext_vector_type(8)));

#define NNODES 20000
#define NEDGES 320000

#define BM 128          // block tile M
#define BN 64           // block tile N
#define BK 32           // K step (one bf16 WMMA)
#define SK 40           // padded LDS row stride (elements); 80B keeps 16B align

// ---------------------------------------------------------------------------
__global__ __launch_bounds__(256)
void fill_f32(float* __restrict__ p, float v, long long n) {
    long long i = (long long)blockIdx.x * blockDim.x + threadIdx.x;
    if (i < n) p[i] = v;
}

__global__ __launch_bounds__(256)
void count_edges(const long long* __restrict__ dst, float* __restrict__ cnt, int E) {
    int e = blockIdx.x * blockDim.x + threadIdx.x;
    if (e < E) atomicAdd(&cnt[(int)dst[e]], 1.0f);
}

__global__ __launch_bounds__(256)
void make_inv(float* __restrict__ cnt, int N) {
    int i = blockIdx.x * blockDim.x + threadIdx.x;
    if (i < N) cnt[i] = 1.0f / fmaxf(cnt[i], 1.0f);
}

// one thread = one (edge, 4-feature chunk): float4 gather + 4 atomic f32 adds
__global__ __launch_bounds__(256)
void scatter_add(const float* __restrict__ h,
                 const long long* __restrict__ src,
                 const long long* __restrict__ dst,
                 float* __restrict__ agg, int d4, long long total) {
    long long idx = (long long)blockIdx.x * blockDim.x + threadIdx.x;
    if (idx >= total) return;
    int e = (int)(idx / d4);
    int f = (int)(idx % d4) * 4;
    int d = d4 * 4;
    int s = (int)src[e];
    int t = (int)dst[e];
    const float4 v = *(const float4*)(h + (size_t)s * d + f);
    float* o = agg + (size_t)t * d + f;
    atomicAdd(o + 0, v.x);
    atomicAdd(o + 1, v.y);
    atomicAdd(o + 2, v.z);
    atomicAdd(o + 3, v.w);
}

// ---------------------------------------------------------------------------
// out[M,Nout] = act( A0*scale0 @ B0 + (A1 ? A1 @ B1 : 0) + bias )
//   A0/A1: [M,K] f32 row-major (A0 optionally row-scaled by scale0 = 1/deg)
//   B0/B1: [K,Nout] f32 row-major
// 8 waves: 4x2 grid, each wave -> 32x32 output via 4 accumulators.
// ---------------------------------------------------------------------------
__global__ __launch_bounds__(256)
void sage_gemm(const float* __restrict__ A0, const float* __restrict__ scale0,
               const float* __restrict__ B0,
               const float* __restrict__ A1, const float* __restrict__ B1,
               const float* __restrict__ bias, float* __restrict__ out,
               int M, int K, int Nout, int relu) {
    __shared__ bf16_t As[BM * SK];   // A tile, row-major [m][k]
    __shared__ bf16_t Bs[BN * SK];   // B tile, transposed [n][k]

    const int tid  = threadIdx.x;
    const int m0   = blockIdx.x * BM;
    const int n0   = blockIdx.y * BN;
    const int wave = tid >> 5;
    const int lane = tid & 31;
    const int wm0  = (wave >> 1) * 32;   // wave M offset in tile
    const int wn0  = (wave & 1) * 32;    // wave N offset in tile
    const int half = lane >> 4;          // lane group (0: lanes 0-15, 1: 16-31)
    const int l15  = lane & 15;

    v8f acc[2][2];
    for (int s = 0; s < 2; ++s)
        for (int t = 0; t < 2; ++t)
            acc[s][t] = (v8f){0.f, 0.f, 0.f, 0.f, 0.f, 0.f, 0.f, 0.f};

    for (int phase = 0; phase < 2; ++phase) {
        const float* A  = phase ? A1 : A0;
        const float* B  = phase ? B1 : B0;
        const float* sc = phase ? nullptr : scale0;
        if (!A) break;

        for (int k0 = 0; k0 < K; k0 += BK) {
            // ---- stage A tile: each thread converts half a row (16 f32 -> bf16)
            {
                int row = tid >> 1;
                int cb  = (tid & 1) * 16;
                int gm  = m0 + row;
                float fv[16];
                if (gm < M) {
                    const float* p = A + (size_t)gm * K + k0 + cb;
                    const float4 f0 = ((const float4*)p)[0];
                    const float4 f1 = ((const float4*)p)[1];
                    const float4 f2 = ((const float4*)p)[2];
                    const float4 f3 = ((const float4*)p)[3];
                    fv[0]=f0.x; fv[1]=f0.y; fv[2]=f0.z; fv[3]=f0.w;
                    fv[4]=f1.x; fv[5]=f1.y; fv[6]=f1.z; fv[7]=f1.w;
                    fv[8]=f2.x; fv[9]=f2.y; fv[10]=f2.z; fv[11]=f2.w;
                    fv[12]=f3.x; fv[13]=f3.y; fv[14]=f3.z; fv[15]=f3.w;
                    if (sc) {
                        float s = sc[gm];
                        #pragma unroll
                        for (int i = 0; i < 16; ++i) fv[i] *= s;
                    }
                } else {
                    #pragma unroll
                    for (int i = 0; i < 16; ++i) fv[i] = 0.f;
                }
                union PK { bf16_t h[8]; uint4 u; } p0, p1;
                #pragma unroll
                for (int i = 0; i < 8; ++i) { p0.h[i] = (bf16_t)fv[i]; p1.h[i] = (bf16_t)fv[8 + i]; }
                *(uint4*)&As[row * SK + cb]     = p0.u;
                *(uint4*)&As[row * SK + cb + 8] = p1.u;
            }
            // ---- stage B tile transposed: coalesced f32 reads, bf16 stores
            #pragma unroll
            for (int r = 0; r < 2; ++r) {
                int idx = tid + r * 256;
                int kk  = idx >> 4;           // 0..31
                int nb  = (idx & 15) * 4;     // 0..60
                const float4 f = *(const float4*)(B + (size_t)(k0 + kk) * Nout + n0 + nb);
                Bs[(nb + 0) * SK + kk] = (bf16_t)f.x;
                Bs[(nb + 1) * SK + kk] = (bf16_t)f.y;
                Bs[(nb + 2) * SK + kk] = (bf16_t)f.z;
                Bs[(nb + 3) * SK + kk] = (bf16_t)f.w;
            }
            __syncthreads();

            // ---- load fragments per CDNA5 ISA layouts, 4 WMMAs per wave
            union FR { uint4 u[2]; v16bf v; };
            v16bf afr[2], bfr[2];
            #pragma unroll
            for (int s = 0; s < 2; ++s) {
                // A 16x32: lane half 0 -> K[0..8)+K[16..24); half 1 -> +8
                const uint4* pa = (const uint4*)&As[(wm0 + s * 16 + l15) * SK + half * 8];
                FR f; f.u[0] = pa[0]; f.u[1] = pa[2];   // +0 and +16 elements
                afr[s] = f.v;
            }
            #pragma unroll
            for (int t = 0; t < 2; ++t) {
                // B 32x16 (stored [n][k]): lane half 0 -> K[0..8)+K[16..24); half 1 -> +8
                const uint4* pb = (const uint4*)&Bs[(wn0 + t * 16 + l15) * SK + half * 8];
                FR f; f.u[0] = pb[0]; f.u[1] = pb[2];
                bfr[t] = f.v;
            }
            #pragma unroll
            for (int s = 0; s < 2; ++s)
                #pragma unroll
                for (int t = 0; t < 2; ++t)
                    acc[s][t] = __builtin_amdgcn_wmma_f32_16x16x32_bf16(
                        false, afr[s], false, bfr[t], (short)0, acc[s][t], false, false);
            __syncthreads();
        }
    }

    // ---- epilogue: C layout VGPR v -> M = v + 8*half, N = lane&15
    #pragma unroll
    for (int s = 0; s < 2; ++s) {
        #pragma unroll
        for (int t = 0; t < 2; ++t) {
            int gc = n0 + wn0 + t * 16 + l15;
            float bv = bias[gc];
            #pragma unroll
            for (int v = 0; v < 8; ++v) {
                int gr = m0 + wm0 + s * 16 + v + 8 * half;
                if (gr < M) {
                    float val = acc[s][t][v] + bv;
                    if (relu) val = fmaxf(val, 0.f);
                    out[(size_t)gr * Nout + gc] = val;
                }
            }
        }
    }
}

// ---------------------------------------------------------------------------
static inline int blocks_for(long long n, int t) { return (int)((n + t - 1) / t); }

extern "C" void kernel_launch(void* const* d_in, const int* in_sizes, int n_in,
                              void* d_out, int out_size, void* d_ws, size_t ws_size,
                              hipStream_t stream) {
    const float*     x   = (const float*)d_in[0];
    const long long* ei  = (const long long*)d_in[1];   // int64 [2, E] flat
    const float*     Wl1 = (const float*)d_in[2];
    const float*     bl1 = (const float*)d_in[3];
    const float*     Wr1 = (const float*)d_in[4];
    const float*     Wl2 = (const float*)d_in[5];
    const float*     bl2 = (const float*)d_in[6];
    const float*     Wr2 = (const float*)d_in[7];
    const float*     Wl3 = (const float*)d_in[8];
    const float*     bl3 = (const float*)d_in[9];
    const float*     Wr3 = (const float*)d_in[10];
    const float*     Wfc = (const float*)d_in[11];
    const float*     bfc = (const float*)d_in[12];
    float*           out = (float*)d_out;

    const long long* src = ei;
    const long long* dst = ei + NEDGES;

    // workspace layout (floats)
    float* ws  = (float*)d_ws;
    float* inv = ws;                                  // 20480 (padded)
    float* agg = inv + 20480;                         // N x 512 (reused, <=512 wide)
    float* hA  = agg + (size_t)NNODES * 512;          // N x 512
    float* hB  = hA  + (size_t)NNODES * 512;          // N x 512
    float* h3  = hB  + (size_t)NNODES * 512;          // N x 1024

    const int T = 256;

    // degree -> 1/max(deg,1), computed once, fused into GEMM A-loads
    fill_f32<<<blocks_for(NNODES, T), T, 0, stream>>>(inv, 0.f, NNODES);
    count_edges<<<blocks_for(NEDGES, T), T, 0, stream>>>(dst, inv, NEDGES);
    make_inv<<<blocks_for(NNODES, T), T, 0, stream>>>(inv, NNODES);

    // ---- layer 1: agg(x) -> h1 = relu(mean@Wl1 + bl1 + x@Wr1)   [256 -> 512]
    {
        long long na = (long long)NNODES * 256;
        fill_f32<<<blocks_for(na, T), T, 0, stream>>>(agg, 0.f, na);
        long long tot = (long long)NEDGES * (256 / 4);
        scatter_add<<<blocks_for(tot, T), T, 0, stream>>>(x, src, dst, agg, 256 / 4, tot);
        dim3 g((NNODES + BM - 1) / BM, 512 / BN);
        sage_gemm<<<g, T, 0, stream>>>(agg, inv, Wl1, x, Wr1, bl1, hA,
                                       NNODES, 256, 512, 1);
    }
    // ---- layer 2: [512 -> 512]
    {
        long long na = (long long)NNODES * 512;
        fill_f32<<<blocks_for(na, T), T, 0, stream>>>(agg, 0.f, na);
        long long tot = (long long)NEDGES * (512 / 4);
        scatter_add<<<blocks_for(tot, T), T, 0, stream>>>(hA, src, dst, agg, 512 / 4, tot);
        dim3 g((NNODES + BM - 1) / BM, 512 / BN);
        sage_gemm<<<g, T, 0, stream>>>(agg, inv, Wl2, hA, Wr2, bl2, hB,
                                       NNODES, 512, 512, 1);
    }
    // ---- layer 3: [512 -> 1024]
    {
        long long na = (long long)NNODES * 512;
        fill_f32<<<blocks_for(na, T), T, 0, stream>>>(agg, 0.f, na);
        long long tot = (long long)NEDGES * (512 / 4);
        scatter_add<<<blocks_for(tot, T), T, 0, stream>>>(hB, src, dst, agg, 512 / 4, tot);
        dim3 g((NNODES + BM - 1) / BM, 1024 / BN);
        sage_gemm<<<g, T, 0, stream>>>(agg, inv, Wl3, hB, Wr3, bl3, h3,
                                       NNODES, 512, 1024, 1);
    }
    // ---- FC: out = h3 @ Wfc + bfc   [1024 -> 512], no relu, single phase
    {
        dim3 g((NNODES + BM - 1) / BM, 512 / BN);
        sage_gemm<<<g, T, 0, stream>>>(h3, nullptr, Wfc, nullptr, nullptr, bfc, out,
                                       NNODES, 1024, 512, 0);
    }
    (void)in_sizes; (void)n_in; (void)out_size; (void)ws_size;
}
